// EdgeNodeGCN_45586782880121
// MI455X (gfx1250) — matrix operand order repaired and checked by reading, not compiled, not run on hardware
//
#include <hip/hip_runtime.h>
#include <hip/hip_bf16.h>
#include <math.h>

typedef __attribute__((ext_vector_type(16))) _Float16 v16h;
typedef __attribute__((ext_vector_type(8)))  _Float16 v8h;
typedef __attribute__((ext_vector_type(8)))  float    v8f;

__device__ __forceinline__ v16h cat8(v8h lo, v8h hi) {
    v16h r;
#pragma unroll
    for (int i = 0; i < 8; ++i) { r[i] = lo[i]; r[i + 8] = hi[i]; }
    return r;
}

__device__ __forceinline__ v8f wmma16(v16h a, v16h b, v8f c) {
    // D = A(16x32 f16) * B(32x16 f16) + C(16x16 f32)
    return __builtin_amdgcn_wmma_f32_16x16x32_f16(false, a, false, b, (short)0, c, false, false);
}

// ---------------------------------------------------------------------------
// Weight swizzle: W[K][N] f32 (row-major) -> per-lane v16h B fragments, f16.
// Fragment (nt, ks): 32 lanes x 16 halves contiguous.
//   lane < 16 : halves h -> W[ks*32 +     (h&7) + 16*(h>>3)][nt*16 + lane]
//   lane >= 16: halves h -> W[ks*32 + 8 + (h&7) + 16*(h>>3)][nt*16 + (lane-16)]
// ---------------------------------------------------------------------------
__global__ void swizzle_weight_f16(const float* __restrict__ W,
                                   _Float16* __restrict__ Wz, int K, int N) {
    int tid = blockIdx.x * blockDim.x + threadIdx.x;
    if (tid >= K * N) return;
    int nK   = K >> 5;                 // K-steps of 32
    int h    = tid & 15;
    int lane = (tid >> 4) & 31;
    int rest = tid >> 9;               // = nt*nK + ks
    int ks   = rest % nK;
    int nt   = rest / nK;
    int k = ks * 32 + ((lane >> 4) << 3) + (h & 7) + ((h >> 3) << 4);
    int n = (nt << 4) + (lane & 15);
    Wz[tid] = (_Float16)W[k * N + n];
}

// ---------------------------------------------------------------------------
// Edge kernel: fused gather + EdgeConv MLP (256->256, WMMA f16) + scatter-add.
// Also folds node path aggregation node_agg[dst] += x[src].
// 4 waves/block, 16 edges per wave.
// ---------------------------------------------------------------------------
#define EK_WAVES 4
#define FSTRIDE  264   // 256 + 8 halves pad (16B-aligned, bank-conflict free)

__global__ void __launch_bounds__(EK_WAVES * 32)
edge_kernel(const float* __restrict__ x,
            const int* __restrict__ edge_index,       // [2][E], src then dst
            const float* __restrict__ b_edge,         // [256]
            const _Float16* __restrict__ WzE,         // swizzled W_edge 256x256
            float* __restrict__ node_agg,             // [N][128]
            float* __restrict__ edge_agg,             // [N][256]
            int E) {
    __shared__ _Float16 Fs[EK_WAVES][16 * FSTRIDE];
    __shared__ int      Dl[EK_WAVES][16];

    const int wave = threadIdx.x >> 5;
    const int lane = threadIdx.x & 31;
    const int row  = lane & 15;
    const int hi   = lane >> 4;
    const int ebase = (blockIdx.x * EK_WAVES + wave) * 16;

    const int* src = edge_index;
    const int* dst = edge_index + E;
    _Float16* F = Fs[wave];

    // ---- phase 1: gather features, build [xi, xj-xi] in LDS (f16),
    //      fold node_agg[dst] += x[src] scatter in the same pass.
    for (int m = 0; m < 16; ++m) {
        int e = ebase + m;
        if (e < E) {
            int s = src[e];
            int d = dst[e];
            if (lane == 0) Dl[wave][m] = d;
            const float* xi = x + (long long)d * 128;
            const float* xj = x + (long long)s * 128;
#pragma unroll
            for (int f = lane; f < 128; f += 32) {
                float a = xi[f];
                float b = xj[f];
                F[m * FSTRIDE + f]       = (_Float16)a;
                F[m * FSTRIDE + 128 + f] = (_Float16)(b - a);
                atomicAdd(node_agg + (long long)d * 128 + f, b);
            }
        } else {
            if (lane == 0) Dl[wave][m] = -1;
            for (int f = lane; f < 256; f += 32) F[m * FSTRIDE + f] = (_Float16)0.f;
        }
    }
    __syncthreads();

    // ---- phase 2: 16x256 = [16x256] @ [256x256] via WMMA, relu+bias,
    //      scatter-add into edge_agg.
    for (int nt = 0; nt < 16; ++nt) {
        v8f acc = {};
#pragma unroll
        for (int ks = 0; ks < 8; ++ks) {
            const _Float16* ap = F + row * FSTRIDE + ks * 32 + hi * 8;
            v16h a = cat8(*(const v8h*)ap, *(const v8h*)(ap + 16));
            v16h b = *(const v16h*)(WzE + (((nt * 8 + ks) * 32 + lane) << 4));
            acc = wmma16(a, b, acc);
        }
        int col = nt * 16 + row;
        float bias = b_edge[col];
#pragma unroll
        for (int r = 0; r < 8; ++r) {
            int m = hi * 8 + r;
            int d = Dl[wave][m];
            float v = acc[r] + bias;
            v = v > 0.f ? v : 0.f;
            if (d >= 0) atomicAdd(edge_agg + (long long)d * 256 + col, v);
        }
    }
}

// ---------------------------------------------------------------------------
// Node kernel: nodes=relu(node_agg@Wn+b), edges=relu(edge_agg@We+b),
// h=relu([nodes,edges]@Wf1+b), out=sigmoid(h@Wf2+b). WMMA for all GEMMs.
// 2 waves/block, 16 nodes per wave.
// ---------------------------------------------------------------------------
#define NK_WAVES 2
#define ASTRIDE  264   // agg staging (max K=256) + pad
#define HSTRIDE  392   // h staging (K=384) + pad

__global__ void __launch_bounds__(NK_WAVES * 32)
node_kernel(const float* __restrict__ node_agg,       // [N][128]
            const float* __restrict__ edge_agg,       // [N][256]
            const _Float16* __restrict__ WzN,         // 128x256 swizzled
            const float* __restrict__ b_node,         // [256]
            const _Float16* __restrict__ WzD,         // 256x128 swizzled
            const float* __restrict__ b_ed,           // [128]
            const _Float16* __restrict__ Wz1,         // 384x32 swizzled
            const float* __restrict__ b_f1,           // [32]
            const float* __restrict__ W_f2,           // [32]
            const float* __restrict__ b_f2,           // [1]
            float* __restrict__ out, int N) {
    __shared__ _Float16 As[NK_WAVES][16 * ASTRIDE];
    __shared__ _Float16 Hs[NK_WAVES][16 * HSTRIDE];

    const int wave = threadIdx.x >> 5;
    const int lane = threadIdx.x & 31;
    const int row  = lane & 15;
    const int hi   = lane >> 4;
    const int nbase = (blockIdx.x * NK_WAVES + wave) * 16;

    _Float16* A = As[wave];
    _Float16* H = Hs[wave];

    // ---- stage node_agg rows (f32 -> f16)
    for (int m = 0; m < 16; ++m) {
        int node = nbase + m; if (node >= N) node = N - 1;
        const float* p = node_agg + (long long)node * 128;
#pragma unroll
        for (int f = lane; f < 128; f += 32) A[m * 136 + f] = (_Float16)p[f];
    }
    __syncthreads();

    // ---- nodes = relu([16x128] @ Wn[128x256] + b_node) -> H cols 0..255
    for (int nt = 0; nt < 16; ++nt) {
        v8f acc = {};
#pragma unroll
        for (int ks = 0; ks < 4; ++ks) {
            const _Float16* ap = A + row * 136 + ks * 32 + hi * 8;
            v16h a = cat8(*(const v8h*)ap, *(const v8h*)(ap + 16));
            v16h b = *(const v16h*)(WzN + (((nt * 4 + ks) * 32 + lane) << 4));
            acc = wmma16(a, b, acc);
        }
        int col = nt * 16 + row;
        float bias = b_node[col];
#pragma unroll
        for (int r = 0; r < 8; ++r) {
            float v = acc[r] + bias;
            H[(hi * 8 + r) * HSTRIDE + col] = (_Float16)(v > 0.f ? v : 0.f);
        }
    }
    __syncthreads();

    // ---- stage edge_agg rows (reuse A arena, stride 264)
    for (int m = 0; m < 16; ++m) {
        int node = nbase + m; if (node >= N) node = N - 1;
        const float* p = edge_agg + (long long)node * 256;
#pragma unroll
        for (int f = lane; f < 256; f += 32) A[m * ASTRIDE + f] = (_Float16)p[f];
    }
    __syncthreads();

    // ---- edges = relu([16x256] @ We[256x128] + b_ed) -> H cols 256..383
    for (int nt = 0; nt < 8; ++nt) {
        v8f acc = {};
#pragma unroll
        for (int ks = 0; ks < 8; ++ks) {
            const _Float16* ap = A + row * ASTRIDE + ks * 32 + hi * 8;
            v16h a = cat8(*(const v8h*)ap, *(const v8h*)(ap + 16));
            v16h b = *(const v16h*)(WzD + (((nt * 8 + ks) * 32 + lane) << 4));
            acc = wmma16(a, b, acc);
        }
        int col = nt * 16 + row;
        float bias = b_ed[col];
#pragma unroll
        for (int r = 0; r < 8; ++r) {
            float v = acc[r] + bias;
            H[(hi * 8 + r) * HSTRIDE + 256 + col] = (_Float16)(v > 0.f ? v : 0.f);
        }
    }
    __syncthreads();

    // ---- h = relu([16x384] @ Wf1[384x32] + b_f1) -> A arena, stride 40
    for (int nt = 0; nt < 2; ++nt) {
        v8f acc = {};
#pragma unroll
        for (int ks = 0; ks < 12; ++ks) {
            const _Float16* ap = H + row * HSTRIDE + ks * 32 + hi * 8;
            v16h a = cat8(*(const v8h*)ap, *(const v8h*)(ap + 16));
            v16h b = *(const v16h*)(Wz1 + (((nt * 12 + ks) * 32 + lane) << 4));
            acc = wmma16(a, b, acc);
        }
        int col = nt * 16 + row;
        float bias = b_f1[col];
#pragma unroll
        for (int r = 0; r < 8; ++r) {
            float v = acc[r] + bias;
            A[(hi * 8 + r) * 40 + col] = (_Float16)(v > 0.f ? v : 0.f);
        }
    }
    __syncthreads();

    // ---- out = sigmoid(h @ Wf2 + b_f2)   (32 -> 1, VALU dot)
    if (lane < 16) {
        int node = nbase + lane;
        if (node < N) {
            float z = b_f2[0];
#pragma unroll
            for (int k = 0; k < 32; ++k) z += (float)A[lane * 40 + k] * W_f2[k];
            out[node] = 1.f / (1.f + expf(-z));
        }
    }
}

// ---------------------------------------------------------------------------
extern "C" void kernel_launch(void* const* d_in, const int* in_sizes, int n_in,
                              void* d_out, int out_size, void* d_ws, size_t ws_size,
                              hipStream_t stream) {
    const float* x       = (const float*)d_in[0];
    const int*   ei      = (const int*)d_in[1];     // [2][E] src,dst (int32)
    // d_in[2] = e : unused by the math
    const float* W_node  = (const float*)d_in[3];
    const float* b_node  = (const float*)d_in[4];
    const float* W_edge  = (const float*)d_in[5];
    const float* b_edge  = (const float*)d_in[6];
    const float* W_ed    = (const float*)d_in[7];
    const float* b_ed    = (const float*)d_in[8];
    const float* W_f1    = (const float*)d_in[9];
    const float* b_f1    = (const float*)d_in[10];
    const float* W_f2    = (const float*)d_in[11];
    const float* b_f2    = (const float*)d_in[12];
    float* out = (float*)d_out;

    const int N = in_sizes[0] / 128;
    const int E = in_sizes[1] / 2;

    // Workspace carve-up (256B aligned)
    auto alignup = [](size_t v) { return (v + 255) & ~(size_t)255; };
    char* ws = (char*)d_ws;
    size_t off = 0;
    float* node_agg = (float*)(ws + off); off += alignup((size_t)N * 128 * sizeof(float));
    float* edge_agg = (float*)(ws + off); off += alignup((size_t)N * 256 * sizeof(float));
    _Float16* WzE = (_Float16*)(ws + off); off += alignup((size_t)256 * 256 * sizeof(_Float16));
    _Float16* WzN = (_Float16*)(ws + off); off += alignup((size_t)128 * 256 * sizeof(_Float16));
    _Float16* WzD = (_Float16*)(ws + off); off += alignup((size_t)256 * 128 * sizeof(_Float16));
    _Float16* Wz1 = (_Float16*)(ws + off); off += alignup((size_t)384 * 32  * sizeof(_Float16));

    // Zero the segment-sum accumulators every call (deterministic launch).
    hipMemsetAsync(node_agg, 0, (size_t)N * 128 * sizeof(float), stream);
    hipMemsetAsync(edge_agg, 0, (size_t)N * 256 * sizeof(float), stream);

    // Swizzle weights into WMMA B-fragment layout (f16).
    auto blocks = [](int n) { return (n + 255) / 256; };
    swizzle_weight_f16<<<blocks(256 * 256), 256, 0, stream>>>(W_edge, WzE, 256, 256);
    swizzle_weight_f16<<<blocks(128 * 256), 256, 0, stream>>>(W_node, WzN, 128, 256);
    swizzle_weight_f16<<<blocks(256 * 128), 256, 0, stream>>>(W_ed,   WzD, 256, 128);
    swizzle_weight_f16<<<blocks(384 * 32),  256, 0, stream>>>(W_f1,   Wz1, 384, 32);

    // Fused edge pass: gather + WMMA MLP + scatter (+ node_agg scatter).
    int egrid = (E + EK_WAVES * 16 - 1) / (EK_WAVES * 16);
    edge_kernel<<<egrid, EK_WAVES * 32, 0, stream>>>(x, ei, b_edge, WzE,
                                                     node_agg, edge_agg, E);

    // Node pass: three WMMA GEMMs + sigmoid head.
    int ngrid = (N + NK_WAVES * 16 - 1) / (NK_WAVES * 16);
    node_kernel<<<ngrid, NK_WAVES * 32, 0, stream>>>(node_agg, edge_agg,
                                                     WzN, b_node, WzD, b_ed,
                                                     Wz1, b_f1, W_f2, b_f2,
                                                     out, N);
}
